// IntraRegionTransformerEncoder_5317169512795
// MI455X (gfx1250) — matrix operand order repaired and checked by reading, not compiled, not run on hardware
//
#include <hip/hip_runtime.h>
#include <math.h>

typedef _Float16 half_t;
typedef __attribute__((ext_vector_type(16))) _Float16 v16h;
typedef __attribute__((ext_vector_type(8)))  float    v8f;

#define WMMA_F16(a, b, c) \
    __builtin_amdgcn_wmma_f32_16x16x32_f16(false, (a), false, (b), (short)0, (c), false, false)

// ---------------- problem constants ----------------
static constexpr int D_MODEL = 256;
static constexpr int PROJ    = 128;
static constexpr int M_TOK_TOTAL = 138240;   // sum nw*occ
static constexpr int S_SLOTS     = 184320;   // sum nw*m_tok
// levels: (n_win, m_tok, occ, slot_base, token_count)
static constexpr int L_NWIN[3] = {1500, 300, 60};
static constexpr int L_MTOK[3] = {64, 192, 512};
static constexpr int L_OCC [3] = {48, 144, 384};
static constexpr int L_SBASE[3]= {0, 96000, 153600};
static constexpr int L_MI  [3] = {72000, 43200, 23040};
// GLT layers: (groups, Kg=in/g, Kpad, Nl=out/g)
// weight f16 offsets (in halves) inside packed glt buffer
static constexpr int GLT_OFF[4] = {0, 12288, 36864, 55296};
static constexpr int GLT_TOTAL_H = 75776;

// ---------------- WMMA fragment helpers ----------------
// A fragment 16x32 f16: lane -> row m = lane&15; comps 0..7 -> K = 8*half + c,
// comps 8..15 -> K = 16 + 8*half + (c-8)   (half = lane>>4)
__device__ __forceinline__ v16h a_frag_row_f16(const half_t* row, int lane) {
    v16h r;
    int h8 = (lane >> 4) * 8;
    if (row) {
#pragma unroll
        for (int c = 0; c < 8; ++c) r[c] = row[h8 + c];
#pragma unroll
        for (int c = 0; c < 8; ++c) r[8 + c] = row[16 + h8 + c];
    } else {
#pragma unroll
        for (int c = 0; c < 16; ++c) r[c] = (half_t)0.0f;
    }
    return r;
}

__device__ __forceinline__ v16h a_frag_row_f32(const float* row, int lane) {
    v16h r;
    int h8 = (lane >> 4) * 8;
#pragma unroll
    for (int c = 0; c < 8; ++c) r[c] = (half_t)row[h8 + c];
#pragma unroll
    for (int c = 0; c < 8; ++c) r[8 + c] = (half_t)row[16 + h8 + c];
    return r;
}

// B fragment 32x16 f16, K-major storage: element (k, n) at p[n*strideK + k].
// lane -> col n = lane&15; comps 0..7 -> K = 8*half + c, comps 8..15 -> K = 16 + 8*half + (c-8)
__device__ __forceinline__ v16h b_frag_kmajor(const half_t* p, int strideK, int lane) {
    int n  = lane & 15;
    int h8 = (lane >> 4) * 8;
    const half_t* col = p + n * strideK;
    v16h r;
#pragma unroll
    for (int c = 0; c < 8; ++c) r[c] = col[h8 + c];
#pragma unroll
    for (int c = 0; c < 8; ++c) r[8 + c] = col[16 + h8 + c];
    return r;
}

__device__ __forceinline__ float gelu_exact(float x) {
    return 0.5f * x * (1.0f + erff(x * 0.70710678118654752f));
}

// ---------------- prep kernels ----------------
__global__ void k_cvt_f16(const float* __restrict__ in, half_t* __restrict__ out, int n) {
    for (int i = blockIdx.x * blockDim.x + threadIdx.x; i < n; i += gridDim.x * blockDim.x)
        out[i] = (half_t)in[i];
}

// glt weight (g, Kg, Nl) -> f16 packed as [grp][o][kpad] (K-major per output col, zero pad)
__global__ void k_cvt_glt(const float* __restrict__ W, half_t* __restrict__ dst,
                          int g, int Kg, int Kpad, int Nl) {
    int n = g * Nl * Kpad;
    for (int i = blockIdx.x * blockDim.x + threadIdx.x; i < n; i += gridDim.x * blockDim.x) {
        int grp = i / (Nl * Kpad);
        int rem = i - grp * (Nl * Kpad);
        int o = rem / Kpad;
        int k = rem - o * Kpad;
        dst[i] = (k < Kg) ? (half_t)W[(grp * Kg + k) * Nl + o] : (half_t)0.0f;
    }
}

__global__ void k_init_slot2tok(int* __restrict__ s2t, int n) {
    for (int i = blockIdx.x * blockDim.x + threadIdx.x; i < n; i += gridDim.x * blockDim.x)
        s2t[i] = -1;
}

__global__ void k_scatter_maps(const int* __restrict__ vox, const int* __restrict__ win,
                               int Mi, int slotbase,
                               int* __restrict__ slot2tok, int* __restrict__ tok2slot) {
    for (int j = blockIdx.x * blockDim.x + threadIdx.x; j < Mi; j += gridDim.x * blockDim.x) {
        int s = slotbase + win[j];
        int t = vox[j];
        slot2tok[s] = t;
        tok2slot[t] = s;
    }
}

// res[vox[j]] = src[vox[j]] + pos_flat[win[j]]
__global__ void k_posadd(const float* __restrict__ src, const float* __restrict__ pos,
                         const int* __restrict__ vox, const int* __restrict__ win,
                         int Mi, float* __restrict__ res) {
    int n = Mi * D_MODEL;
    for (int i = blockIdx.x * blockDim.x + threadIdx.x; i < n; i += gridDim.x * blockDim.x) {
        int j = i >> 8;
        int c = i & 255;
        size_t t = (size_t)vox[j] * D_MODEL + c;
        res[t] = src[t] + pos[(size_t)win[j] * D_MODEL + c];
    }
}

// ---------------- DExTra (fused, one wave per 16-token tile) ----------------
struct DextraArgs {
    const float* res;        // (M, 256) f32
    const half_t* w_in16;    // (128, 256) K-major
    const float* b_in;
    const half_t* wl[4];     // packed glt f16
    const float* gbias[4];
    const float* ggamma[4];
    const float* gbeta[4];
    half_t* dx16;            // (M, 128) out
};

template <int G, int KG, int KPAD, int NL>
__device__ __forceinline__ void glt_layer(int lane, const half_t* __restrict__ wl,
                                          const float* __restrict__ bias,
                                          const float* __restrict__ gamma,
                                          const float* __restrict__ beta,
                                          half_t (*act)[512], const half_t (*xin)[128],
                                          float (*outf)[256]) {
    constexpr int CO = G * NL;
    __builtin_prefetch(wl, 0, 1);
    int n = lane & 15;
    int hh = lane >> 4;
#pragma unroll
    for (int grp = 0; grp < G; ++grp) {
#pragma unroll
        for (int nt = 0; nt < NL / 16; ++nt) {
            v8f acc = {};
#pragma unroll
            for (int kt = 0; kt < KPAD / 32; ++kt) {
                v16h af = a_frag_row_f16(&act[lane & 15][grp * KG + kt * 32], lane);
                v16h bf = b_frag_kmajor(wl + (grp * NL + nt * 16) * KPAD + kt * 32, KPAD, lane);
                acc = WMMA_F16(af, bf, acc);
            }
            int olocal = nt * 16 + n;
            float bsv = bias[grp * NL + olocal];
            int fin = olocal * G + grp;  // glt shuffle: interleave groups
#pragma unroll
            for (int r = 0; r < 8; ++r) {
                int m = r + 8 * hh;
                outf[m][fin] = acc[r] + bsv;
            }
        }
    }
    __syncthreads();
    // LayerNorm + exact GELU -> act[0..CO), then append x_in at [CO..CO+128)
    int row = lane & 15;
    constexpr int HCO = CO / 2;
    int c0 = hh * HCO;
    float s = 0.f, s2 = 0.f;
    for (int c = 0; c < HCO; ++c) {
        float v = outf[row][c0 + c];
        s += v; s2 += v * v;
    }
    s  += __shfl_xor(s, 16, 32);
    s2 += __shfl_xor(s2, 16, 32);
    float mu = s / CO;
    float var = s2 / CO - mu * mu;
    float rstd = rsqrtf(var + 1e-5f);
    for (int c = 0; c < HCO; ++c) {
        int ch = c0 + c;
        float v = (outf[row][ch] - mu) * rstd * gamma[ch] + beta[ch];
        act[row][ch] = (half_t)gelu_exact(v);
    }
    for (int c = hh * 64; c < hh * 64 + 64; ++c) act[row][CO + c] = xin[row][c];
    __syncthreads();
}

__global__ __launch_bounds__(32) void k_dextra(DextraArgs args) {
    __shared__ half_t act[16][512];
    __shared__ half_t xin[16][128];
    __shared__ float  outf[16][256];
    int lane = threadIdx.x;
    int t0 = blockIdx.x * 16;

    for (int i = lane; i < 16 * 512; i += 32) ((half_t*)act)[i] = (half_t)0.0f;
    __syncthreads();

    // x_in = res @ w_in^T + b_in  : (16,256) x (256,128)
    const float* arow = args.res + (size_t)(t0 + (lane & 15)) * D_MODEL;
    v16h a[8];
#pragma unroll
    for (int kt = 0; kt < 8; ++kt) a[kt] = a_frag_row_f32(arow + kt * 32, lane);
    int n = lane & 15, hh = lane >> 4;
#pragma unroll
    for (int nt = 0; nt < 8; ++nt) {
        v8f acc = {};
#pragma unroll
        for (int kt = 0; kt < 8; ++kt) {
            v16h bf = b_frag_kmajor(args.w_in16 + (nt * 16) * 256 + kt * 32, 256, lane);
            acc = WMMA_F16(a[kt], bf, acc);
        }
        int col = nt * 16 + n;
        float bias = args.b_in[col];
#pragma unroll
        for (int r = 0; r < 8; ++r) {
            int m = r + 8 * hh;
            half_t hv = (half_t)(acc[r] + bias);
            act[m][col] = hv;
            xin[m][col] = hv;
        }
    }
    __syncthreads();

    glt_layer<2,  64,  64, 96>(lane, args.wl[0], args.gbias[0], args.ggamma[0], args.gbeta[0], act, xin, outf);
    glt_layer<4,  80,  96, 64>(lane, args.wl[1], args.gbias[1], args.ggamma[1], args.gbeta[1], act, xin, outf);
    glt_layer<4,  96,  96, 48>(lane, args.wl[2], args.gbias[2], args.ggamma[2], args.gbeta[2], act, xin, outf);
    glt_layer<2, 160, 160, 64>(lane, args.wl[3], args.gbias[3], args.ggamma[3], args.gbeta[3], act, xin, outf);

    for (int i = lane; i < 16 * 128; i += 32) {
        int m = i >> 7, c = i & 127;
        args.dx16[(size_t)(t0 + m) * PROJ + c] = act[m][c];
    }
}

// ---------------- QKV projection over all window slots ----------------
struct QKVArgs {
    const half_t* dx16;      // (M,128)
    const int* slot2tok;     // (S)
    const half_t* wq16; const half_t* wk16; const half_t* wv16;
    const float* bq; const float* bk; const float* bv;
    half_t* q16;             // (S,128) row-major, scaled, +bq
    half_t* k16;             // (S,128) row-major, +bk
    half_t* vT16;            // (128,S) transposed, +bv
};

__global__ __launch_bounds__(32) void k_qkv(QKVArgs args) {
    int lane = threadIdx.x;
    int s0 = blockIdx.x * 16;
    int n = lane & 15, hh = lane >> 4;
    int t = args.slot2tok[s0 + n];  // A-frag row m == lane&15
    const half_t* arow = (t >= 0) ? (args.dx16 + (size_t)t * PROJ) : (const half_t*)0;
    v16h a[4];
#pragma unroll
    for (int kt = 0; kt < 4; ++kt)
        a[kt] = a_frag_row_f16(arow ? arow + kt * 32 : (const half_t*)0, lane);

    const float scale = 0.088388347648318447f;  // 128^-0.5
#pragma unroll
    for (int nt = 0; nt < 8; ++nt) {
        int col = nt * 16 + n;
        // Q
        v8f acc = {};
#pragma unroll
        for (int kt = 0; kt < 4; ++kt)
            acc = WMMA_F16(a[kt], b_frag_kmajor(args.wq16 + (nt * 16) * 128 + kt * 32, 128, lane), acc);
        float bq = args.bq[col];
#pragma unroll
        for (int r = 0; r < 8; ++r) {
            int m = r + 8 * hh;
            args.q16[(size_t)(s0 + m) * PROJ + col] = (half_t)((acc[r] + bq) * scale);
        }
        // K
        v8f acck = {};
#pragma unroll
        for (int kt = 0; kt < 4; ++kt)
            acck = WMMA_F16(a[kt], b_frag_kmajor(args.wk16 + (nt * 16) * 128 + kt * 32, 128, lane), acck);
        float bk = args.bk[col];
#pragma unroll
        for (int r = 0; r < 8; ++r) {
            int m = r + 8 * hh;
            args.k16[(size_t)(s0 + m) * PROJ + col] = (half_t)(acck[r] + bk);
        }
        // V (stored transposed)
        v8f accv = {};
#pragma unroll
        for (int kt = 0; kt < 4; ++kt)
            accv = WMMA_F16(a[kt], b_frag_kmajor(args.wv16 + (nt * 16) * 128 + kt * 32, 128, lane), accv);
        float bv = args.bv[col];
#pragma unroll
        for (int r = 0; r < 8; ++r) {
            int m = r + 8 * hh;
            args.vT16[(size_t)col * S_SLOTS + (s0 + m)] = (half_t)(accv[r] + bv);
        }
    }
}

// ---------------- per-window attention (dense over occupied prefix) ----------------
struct AttnArgs {
    const half_t* q16; const half_t* k16; const half_t* vT16;
    half_t* oav16;     // (S,128)
    int slotbase; int mtok; int occ; int qtiles;
};

__global__ __launch_bounds__(32) void k_attn(AttnArgs args) {
    __shared__ float  sc[16][384];
    __shared__ half_t pb[16][384];
    int lane = threadIdx.x;
    int win = blockIdx.x / args.qtiles;
    int qt  = blockIdx.x - win * args.qtiles;
    int sb  = args.slotbase + win * args.mtok;
    int q0  = sb + qt * 16;
    int n = lane & 15, hh = lane >> 4;
    int occ = args.occ;

    const half_t* qrow = args.q16 + (size_t)(q0 + n) * PROJ;
    v16h a[4];
#pragma unroll
    for (int kt = 0; kt < 4; ++kt) a[kt] = a_frag_row_f16(qrow + kt * 32, lane);

    // scores S = Q K^T over occupied keys
    for (int nt = 0; nt < occ / 16; ++nt) {
        v8f acc = {};
#pragma unroll
        for (int kt = 0; kt < 4; ++kt) {
            v16h bf = b_frag_kmajor(args.k16 + (size_t)(sb + nt * 16) * PROJ + kt * 32, PROJ, lane);
            acc = WMMA_F16(a[kt], bf, acc);
        }
#pragma unroll
        for (int r = 0; r < 8; ++r) sc[r + 8 * hh][nt * 16 + n] = acc[r];
    }
    __syncthreads();

    // row softmax over [0, occ)
    int row = lane & 15;
    int hocc = occ / 2;
    int c0 = hh * hocc;
    float mx = -3.0e38f;
    for (int c = 0; c < hocc; ++c) mx = fmaxf(mx, sc[row][c0 + c]);
    mx = fmaxf(mx, __shfl_xor(mx, 16, 32));
    float sum = 0.f;
    for (int c = 0; c < hocc; ++c) {
        float e = __expf(sc[row][c0 + c] - mx);
        sc[row][c0 + c] = e;
        sum += e;
    }
    sum += __shfl_xor(sum, 16, 32);
    float rcp = 1.0f / sum;
    for (int c = 0; c < hocc; ++c) pb[row][c0 + c] = (half_t)(sc[row][c0 + c] * rcp);
    int pocc = (occ + 31) & ~31;
    int padw = pocc - occ;
    if (padw > 0) {
        for (int i = lane; i < 16 * padw; i += 32)
            pb[i / padw][occ + (i % padw)] = (half_t)0.0f;
    }
    __syncthreads();

    // O = P V  (V transposed in global)
#pragma unroll
    for (int nt = 0; nt < 8; ++nt) {
        v8f acc = {};
        for (int kt = 0; kt < pocc / 32; ++kt) {
            v16h af = a_frag_row_f16(&pb[lane & 15][kt * 32], lane);
            v16h bf = b_frag_kmajor(args.vT16 + (size_t)(nt * 16) * S_SLOTS + sb + kt * 32,
                                    S_SLOTS, lane);
            acc = WMMA_F16(af, bf, acc);
        }
#pragma unroll
        for (int r = 0; r < 8; ++r) {
            int m = r + 8 * hh;
            args.oav16[(size_t)(q0 + m) * PROJ + nt * 16 + n] = (half_t)acc[r];
        }
    }
}

// ---------------- output proj + residual + LN1 + FFN + LN2 ----------------
struct FinalArgs {
    const half_t* oav16;   // (S,128)
    const int* tok2slot;   // (M)
    const half_t* wo16; const float* bo;
    const float* resf;     // (M,256) residual
    const float* ln1g; const float* ln1b;
    const half_t* fc1_16; const float* fc1b;
    const half_t* fc2_16; const float* fc2b;
    const float* ln2g; const float* ln2b;
    float* out;            // (M,256)
};

__global__ __launch_bounds__(32) void k_final(FinalArgs args) {
    __shared__ float  xf[16][256];
    __shared__ half_t xh[16][256];
    __shared__ half_t hb[16][64];
    int lane = threadIdx.x;
    int t0 = blockIdx.x * 16;
    int n = lane & 15, hh = lane >> 4;

    int slot = args.tok2slot[t0 + n];
    const half_t* arow = args.oav16 + (size_t)slot * PROJ;
    v16h a[4];
#pragma unroll
    for (int kt = 0; kt < 4; ++kt) a[kt] = a_frag_row_f16(arow + kt * 32, lane);

    // o @ wo^T + bo + residual
#pragma unroll
    for (int nt = 0; nt < 16; ++nt) {
        v8f acc = {};
#pragma unroll
        for (int kt = 0; kt < 4; ++kt)
            acc = WMMA_F16(a[kt], b_frag_kmajor(args.wo16 + (nt * 16) * 128 + kt * 32, 128, lane), acc);
        int col = nt * 16 + n;
        float bo = args.bo[col];
#pragma unroll
        for (int r = 0; r < 8; ++r) {
            int m = r + 8 * hh;
            xf[m][col] = acc[r] + bo + args.resf[(size_t)(t0 + m) * D_MODEL + col];
        }
    }
    __syncthreads();

    // LN1
    int row = lane & 15;
    {
        int c0 = hh * 128;
        float s = 0.f, s2 = 0.f;
        for (int c = 0; c < 128; ++c) { float v = xf[row][c0 + c]; s += v; s2 += v * v; }
        s  += __shfl_xor(s, 16, 32);
        s2 += __shfl_xor(s2, 16, 32);
        float mu = s / 256.0f;
        float rstd = rsqrtf(s2 / 256.0f - mu * mu + 1e-5f);
        for (int c = 0; c < 128; ++c) {
            int ch = c0 + c;
            float v = (xf[row][ch] - mu) * rstd * args.ln1g[ch] + args.ln1b[ch];
            xf[row][ch] = v;
            xh[row][ch] = (half_t)v;
        }
    }
    __syncthreads();

    // FFN up: (16,256) x (256,64), exact GELU
    v16h ax[8];
#pragma unroll
    for (int kt = 0; kt < 8; ++kt) ax[kt] = a_frag_row_f16(&xh[lane & 15][kt * 32], lane);
#pragma unroll
    for (int nt = 0; nt < 4; ++nt) {
        v8f acc = {};
#pragma unroll
        for (int kt = 0; kt < 8; ++kt)
            acc = WMMA_F16(ax[kt], b_frag_kmajor(args.fc1_16 + (nt * 16) * 256 + kt * 32, 256, lane), acc);
        int col = nt * 16 + n;
        float b1 = args.fc1b[col];
#pragma unroll
        for (int r = 0; r < 8; ++r) {
            int m = r + 8 * hh;
            hb[m][col] = (half_t)gelu_exact(acc[r] + b1);
        }
    }
    __syncthreads();

    // FFN down: (16,64) x (64,256) + residual, into xf
    v16h ah[2];
#pragma unroll
    for (int kt = 0; kt < 2; ++kt) ah[kt] = a_frag_row_f16(&hb[lane & 15][kt * 32], lane);
#pragma unroll
    for (int nt = 0; nt < 16; ++nt) {
        v8f acc = {};
#pragma unroll
        for (int kt = 0; kt < 2; ++kt)
            acc = WMMA_F16(ah[kt], b_frag_kmajor(args.fc2_16 + (nt * 16) * 64 + kt * 32, 64, lane), acc);
        int col = nt * 16 + n;
        float b2 = args.fc2b[col];
#pragma unroll
        for (int r = 0; r < 8; ++r) {
            int m = r + 8 * hh;
            xf[m][col] = acc[r] + b2 + xf[m][col];
        }
    }
    __syncthreads();

    // LN2 -> output
    {
        int c0 = hh * 128;
        float s = 0.f, s2 = 0.f;
        for (int c = 0; c < 128; ++c) { float v = xf[row][c0 + c]; s += v; s2 += v * v; }
        s  += __shfl_xor(s, 16, 32);
        s2 += __shfl_xor(s2, 16, 32);
        float mu = s / 256.0f;
        float rstd = rsqrtf(s2 / 256.0f - mu * mu + 1e-5f);
        for (int c = 0; c < 128; ++c) {
            int ch = c0 + c;
            args.out[(size_t)(t0 + row) * D_MODEL + ch] =
                (xf[row][ch] - mu) * rstd * args.ln2g[ch] + args.ln2b[ch];
        }
    }
}

// ---------------- host launcher ----------------
extern "C" void kernel_launch(void* const* d_in, const int* in_sizes, int n_in,
                              void* d_out, int out_size, void* d_ws, size_t ws_size,
                              hipStream_t stream) {
    (void)n_in; (void)out_size; (void)ws_size;
    // ---- parse inputs (setup_inputs dict order; lists handled concat-or-split) ----
    int idx = 0;
    const float* src = (const float*)d_in[idx++];
    const int* vox[3]; const int* win[3]; const float* pos[3];
    for (int i = 0; i < 3; ++i) {
        vox[i] = (const int*)d_in[idx++];
        win[i] = (const int*)d_in[idx++];
        idx++;  // mask (unused; static prefix occupancy)
        pos[i] = (const float*)d_in[idx++];
    }
    const float* w_in = (const float*)d_in[idx++];
    const float* b_in = (const float*)d_in[idx++];

    const float* gw[4];
    if (in_sizes[idx] == 71680) {
        const float* b = (const float*)d_in[idx++];
        gw[0] = b; gw[1] = b + 12288; gw[2] = b + 12288 + 20480; gw[3] = b + 12288 + 20480 + 18432;
    } else {
        for (int i = 0; i < 4; ++i) gw[i] = (const float*)d_in[idx++];
    }
    static const int co_sz[4] = {192, 256, 192, 128};
    const float* gb[4]; const float* gg[4]; const float* gbt[4];
    for (int L = 0; L < 3; ++L) {
        const float** dst = (L == 0) ? gb : (L == 1) ? gg : gbt;
        if (in_sizes[idx] == 768) {
            const float* b = (const float*)d_in[idx++];
            int off = 0;
            for (int i = 0; i < 4; ++i) { dst[i] = b + off; off += co_sz[i]; }
        } else {
            for (int i = 0; i < 4; ++i) dst[i] = (const float*)d_in[idx++];
        }
    }
    const float* wq = (const float*)d_in[idx++];
    const float* wk = (const float*)d_in[idx++];
    const float* wv = (const float*)d_in[idx++];
    const float* bq = (const float*)d_in[idx++];
    const float* bk = (const float*)d_in[idx++];
    const float* bv = (const float*)d_in[idx++];
    const float* wo = (const float*)d_in[idx++];
    const float* bo = (const float*)d_in[idx++];
    const float* ln1g = (const float*)d_in[idx++];
    const float* ln1b = (const float*)d_in[idx++];
    const float* fc1w = (const float*)d_in[idx++];
    const float* fc1b = (const float*)d_in[idx++];
    const float* fc2w = (const float*)d_in[idx++];
    const float* fc2b = (const float*)d_in[idx++];
    const float* ln2g = (const float*)d_in[idx++];
    const float* ln2b = (const float*)d_in[idx++];

    // ---- workspace carve-out ----
    size_t off = 0;
    char* ws = (char*)d_ws;
    auto carve = [&](size_t bytes) -> char* {
        char* p = ws + off;
        off = (off + bytes + 255) & ~(size_t)255;
        return p;
    };
    float*  resf   = (float*) carve((size_t)M_TOK_TOTAL * D_MODEL * 4);
    half_t* dx16   = (half_t*)carve((size_t)M_TOK_TOTAL * PROJ * 2);
    half_t* q16    = (half_t*)carve((size_t)S_SLOTS * PROJ * 2);
    half_t* k16    = (half_t*)carve((size_t)S_SLOTS * PROJ * 2);
    half_t* vT16   = (half_t*)carve((size_t)S_SLOTS * PROJ * 2);
    half_t* oav16  = (half_t*)carve((size_t)S_SLOTS * PROJ * 2);
    int*    s2t    = (int*)   carve((size_t)S_SLOTS * 4);
    int*    t2s    = (int*)   carve((size_t)M_TOK_TOTAL * 4);
    half_t* w_in16 = (half_t*)carve(32768 * 2);
    half_t* glt16  = (half_t*)carve((size_t)GLT_TOTAL_H * 2);
    half_t* wq16   = (half_t*)carve(16384 * 2);
    half_t* wk16   = (half_t*)carve(16384 * 2);
    half_t* wv16   = (half_t*)carve(16384 * 2);
    half_t* wo16   = (half_t*)carve(32768 * 2);
    half_t* fc116  = (half_t*)carve(16384 * 2);
    half_t* fc216  = (half_t*)carve(16384 * 2);

    // ---- weight conversions ----
    k_cvt_f16<<<64, 256, 0, stream>>>(w_in, w_in16, 32768);
    k_cvt_f16<<<64, 256, 0, stream>>>(wq, wq16, 16384);
    k_cvt_f16<<<64, 256, 0, stream>>>(wk, wk16, 16384);
    k_cvt_f16<<<64, 256, 0, stream>>>(wv, wv16, 16384);
    k_cvt_f16<<<64, 256, 0, stream>>>(wo, wo16, 32768);
    k_cvt_f16<<<64, 256, 0, stream>>>(fc1w, fc116, 16384);
    k_cvt_f16<<<64, 256, 0, stream>>>(fc2w, fc216, 16384);
    k_cvt_glt<<<64, 256, 0, stream>>>(gw[0], glt16 + GLT_OFF[0], 2, 64, 64, 96);
    k_cvt_glt<<<64, 256, 0, stream>>>(gw[1], glt16 + GLT_OFF[1], 4, 80, 96, 64);
    k_cvt_glt<<<64, 256, 0, stream>>>(gw[2], glt16 + GLT_OFF[2], 4, 96, 96, 48);
    k_cvt_glt<<<64, 256, 0, stream>>>(gw[3], glt16 + GLT_OFF[3], 2, 160, 160, 64);

    // ---- index maps + positional add ----
    k_init_slot2tok<<<(S_SLOTS + 255) / 256, 256, 0, stream>>>(s2t, S_SLOTS);
    for (int i = 0; i < 3; ++i)
        k_scatter_maps<<<(L_MI[i] + 255) / 256, 256, 0, stream>>>(vox[i], win[i], L_MI[i],
                                                                  L_SBASE[i], s2t, t2s);
    for (int i = 0; i < 3; ++i)
        k_posadd<<<(L_MI[i] * D_MODEL + 255) / 256, 256, 0, stream>>>(src, pos[i], vox[i],
                                                                      win[i], L_MI[i], resf);

    // ---- DExTra ----
    DextraArgs da;
    da.res = resf; da.w_in16 = w_in16; da.b_in = b_in;
    for (int i = 0; i < 4; ++i) {
        da.wl[i] = glt16 + GLT_OFF[i];
        da.gbias[i] = gb[i]; da.ggamma[i] = gg[i]; da.gbeta[i] = gbt[i];
    }
    da.dx16 = dx16;
    k_dextra<<<M_TOK_TOTAL / 16, 32, 0, stream>>>(da);

    // ---- QKV over all window slots ----
    QKVArgs qa;
    qa.dx16 = dx16; qa.slot2tok = s2t;
    qa.wq16 = wq16; qa.wk16 = wk16; qa.wv16 = wv16;
    qa.bq = bq; qa.bk = bk; qa.bv = bv;
    qa.q16 = q16; qa.k16 = k16; qa.vT16 = vT16;
    k_qkv<<<S_SLOTS / 16, 32, 0, stream>>>(qa);

    // ---- attention per level ----
    for (int i = 0; i < 3; ++i) {
        AttnArgs aa;
        aa.q16 = q16; aa.k16 = k16; aa.vT16 = vT16; aa.oav16 = oav16;
        aa.slotbase = L_SBASE[i]; aa.mtok = L_MTOK[i]; aa.occ = L_OCC[i];
        aa.qtiles = L_OCC[i] / 16;
        k_attn<<<L_NWIN[i] * aa.qtiles, 32, 0, stream>>>(aa);
    }

    // ---- output projection + residual + LN1 + FFN + LN2 ----
    FinalArgs fa;
    fa.oav16 = oav16; fa.tok2slot = t2s; fa.wo16 = wo16; fa.bo = bo;
    fa.resf = resf; fa.ln1g = ln1g; fa.ln1b = ln1b;
    fa.fc1_16 = fc116; fa.fc1b = fc1b; fa.fc2_16 = fc216; fa.fc2b = fc2b;
    fa.ln2g = ln2g; fa.ln2b = ln2b; fa.out = (float*)d_out;
    k_final<<<M_TOK_TOTAL / 16, 32, 0, stream>>>(fa);
}
